// EncoderSimilarity_17592186044569
// MI455X (gfx1250) — matrix operand order repaired and checked by reading, not compiled, not run on hardware
//
#include <hip/hip_runtime.h>

typedef __attribute__((ext_vector_type(16))) _Float16 v16h;
typedef __attribute__((ext_vector_type(8)))  float    v8f;
typedef _Float16 h8v __attribute__((ext_vector_type(8)));

#define WMMAF16(A, B, C) \
  __builtin_amdgcn_wmma_f32_16x16x32_f16(false, (A), false, (B), (short)0, (C), false, false)

#if __has_builtin(__builtin_amdgcn_global_load_async_to_lds_b128) && \
    __has_builtin(__builtin_amdgcn_s_wait_asynccnt)
#define HAVE_ASYNC_LDS 1
#else
#define HAVE_ASYNC_LDS 0
#endif

// Async-copy operand types per the builtin signature: v4i pointers in AS1 / AS3.
typedef int i4v __attribute__((vector_size(16)));
typedef __attribute__((address_space(1))) i4v gbl_i4v;
typedef __attribute__((address_space(3))) i4v lds_i4v;

// Generic->global: numerically identical 64-bit address.
__device__ __forceinline__ gbl_i4v* gcast(const void* p) {
  return (gbl_i4v*)(unsigned long long)p;
}
// Generic->LDS: low 32 bits of a generic pointer to __shared__ are the LDS byte offset.
__device__ __forceinline__ lds_i4v* lcast(void* p) {
  return (lds_i4v*)(unsigned int)(unsigned long long)p;
}

// ---------------------------------------------------------------------------
// WMMA fragment loaders (wave32, f16 16x16x32)
// A (16xK tile, row-major, ld in halves): lane m=lane&15; lanes<16 hold K 0-15;
// lanes>=16 hold K 16-31 (contiguous, matching B).
// B (NxK row-major, i.e. 32x16 K-major operand): lane n=lane&15, K-range (lane>>4)*16.
// C/D: element r <-> (row = (lane>>4)*8 + r, col = lane&15).
// ---------------------------------------------------------------------------
__device__ __forceinline__ v16h load_a16(const _Float16* s, int ld) {
  const int lane = threadIdx.x & 31;
  const int m = lane & 15;
  const int kb = (lane & 16) ? 16 : 0;
  const _Float16* p = s + m * ld + kb;
  v16h a;
#pragma unroll
  for (int h = 0; h < 8; ++h) a[h] = p[h];
#pragma unroll
  for (int h = 0; h < 8; ++h) a[h + 8] = p[8 + h];
  return a;
}

__device__ __forceinline__ v16h load_b16(const _Float16* p0, int ld) {
  const int lane = threadIdx.x & 31;
  const int n = lane & 15;
  const int kb = (lane & 16) ? 16 : 0;
  const _Float16* p = p0 + (size_t)n * ld + kb;
  v16h b;
#pragma unroll
  for (int h = 0; h < 16; ++h) b[h] = p[h];
  return b;
}

// ---------------------------------------------------------------------------
// Generic GEMM: C[MxN] = A[MxK] * B[NxK]^T (f16 in, f32 out). M%64==0, N%128==0, K%32==0.
// Double-buffered; async global->LDS copies (ASYNCcnt pipelined) when available.
// ---------------------------------------------------------------------------
__global__ __launch_bounds__(256) void gemm_f16_nt(const _Float16* __restrict__ A,
                                                   const _Float16* __restrict__ Bm,
                                                   float* __restrict__ C,
                                                   int M, int N, int K) {
  __shared__ _Float16 sA[2][64 * 32];
  __shared__ _Float16 sB[2][128 * 32];
  const int tid = threadIdx.x;
  const int wave = tid >> 5, lane = tid & 31;
  const int bm = blockIdx.y * 64, bn = blockIdx.x * 128;
  const int wm = (wave & 1) * 32, wn = (wave >> 1) * 32;
  const int ra = tid >> 2, ca = (tid & 3) * 8;    // A: 64x32 halves, 16B per thread
  const int rb2 = tid >> 1, cb2 = (tid & 1) * 16; // B: 128x32 halves, 32B per thread
  v8f acc[2][2] = {};
  const int KT = K >> 5;

#if HAVE_ASYNC_LDS
  auto stage = [&](int buf, int k0) {
    __builtin_amdgcn_global_load_async_to_lds_b128(
        gcast(&A[(size_t)(bm + ra) * K + k0 + ca]),
        lcast(&sA[buf][ra * 32 + ca]), 0, 0);
    __builtin_amdgcn_global_load_async_to_lds_b128(
        gcast(&Bm[(size_t)(bn + rb2) * K + k0 + cb2]),
        lcast(&sB[buf][rb2 * 32 + cb2]), 0, 0);
    __builtin_amdgcn_global_load_async_to_lds_b128(
        gcast(&Bm[(size_t)(bn + rb2) * K + k0 + cb2 + 8]),
        lcast(&sB[buf][rb2 * 32 + cb2 + 8]), 0, 0);
  };
#else
  auto stage = [&](int buf, int k0) {
    *(h8v*)&sA[buf][ra * 32 + ca] = *(const h8v*)&A[(size_t)(bm + ra) * K + k0 + ca];
    *(h8v*)&sB[buf][rb2 * 32 + cb2] = *(const h8v*)&Bm[(size_t)(bn + rb2) * K + k0 + cb2];
    *(h8v*)&sB[buf][rb2 * 32 + cb2 + 8] =
        *(const h8v*)&Bm[(size_t)(bn + rb2) * K + k0 + cb2 + 8];
  };
#endif

  stage(0, 0);
  for (int kt = 0; kt < KT; ++kt) {
    const int cur = kt & 1;
    const bool more = (kt + 1) < KT;
    if (more) stage(cur ^ 1, (kt + 1) << 5);  // prefetch next tile (overlaps compute)
#if HAVE_ASYNC_LDS
    if (more)
      __builtin_amdgcn_s_wait_asynccnt(3);  // in-order: current tile's 3 ops have landed
    else
      __builtin_amdgcn_s_wait_asynccnt(0);
#endif
    __syncthreads();
    v16h af[2], bf[2];
#pragma unroll
    for (int x = 0; x < 2; ++x) af[x] = load_a16(&sA[cur][(wm + 16 * x) * 32], 32);
#pragma unroll
    for (int y = 0; y < 2; ++y) bf[y] = load_b16(&sB[cur][(wn + 16 * y) * 32], 32);
#pragma unroll
    for (int x = 0; x < 2; ++x)
#pragma unroll
      for (int y = 0; y < 2; ++y) acc[x][y] = WMMAF16(af[x], bf[y], acc[x][y]);
    __syncthreads();  // all waves done reading 'cur' before it is async-overwritten
  }
  const int col = lane & 15, rb = (lane >> 4) * 8;
#pragma unroll
  for (int x = 0; x < 2; ++x)
#pragma unroll
    for (int y = 0; y < 2; ++y)
#pragma unroll
      for (int r = 0; r < 8; ++r)
        C[(size_t)(bm + wm + 16 * x + rb + r) * N + bn + wn + 16 * y + col] = acc[x][y][r];
}

// ---------------------------------------------------------------------------
// Elementwise helpers
// ---------------------------------------------------------------------------
__global__ void f32_to_f16(const float* __restrict__ s, _Float16* __restrict__ d, int n) {
  int i = blockIdx.x * 256 + threadIdx.x;
  if (i < n) d[i] = (_Float16)s[i];
}

__global__ void mean_g(const float* __restrict__ s, _Float16* __restrict__ d16, int G) {
  int i = blockIdx.x * 256 + threadIdx.x;  // over 128*1024
  if (i >= 128 * 1024) return;
  int b = i >> 10, e = i & 1023;
  float acc = 0.f;
  for (int g = 0; g < G; ++g) acc += s[((size_t)b * G + g) * 1024 + e];
  d16[i] = (_Float16)(acc / (float)G);
}

// sim_lm: leaky-relu dot, softmax(lambda*s) over r, weighted sum
__global__ __launch_bounds__(256) void simlm_kernel(const float* __restrict__ Qd,
                                                    float* __restrict__ out0) {
  int idx = blockIdx.x * 256 + threadIdx.x;
  if (idx >= 128 * 128 * 10) return;
  int t = idx % 10;
  int j = (idx / 10) & 127;
  int i = idx / 1280;
  const float* q = Qd + (size_t)(j * 10 + t) * 4608 + i * 36;
  float s[36];
  float mx = -1e30f;
#pragma unroll
  for (int r = 0; r < 36; ++r) {
    float x = q[r];
    x = (x > 0.f) ? x : 0.1f * x;
    s[r] = x;
    mx = fmaxf(mx, x);
  }
  float num = 0.f, den = 0.f;
#pragma unroll
  for (int r = 0; r < 36; ++r) {
    float e = expf(9.0f * (s[r] - mx));
    den += e;
    num += s[r] * e;
  }
  out0[idx] = num / den;
}

// SA epilogue (VisualSA with BN, TextSA without): tanh/softmax/weighted-sum/l2norm
__global__ __launch_bounds__(256) void sa_kernel(
    const float* __restrict__ local, const float* __restrict__ lpre,
    const float* __restrict__ gpre, const float* __restrict__ lb,
    const float* __restrict__ gb, const float* __restrict__ bnlg,
    const float* __restrict__ bnlb, const float* __restrict__ bngg,
    const float* __restrict__ bngb, const float* __restrict__ cw,
    const float* __restrict__ cb, float* __restrict__ out, int G, int hasBN) {
  const int b = blockIdx.x, tid = threadIdx.x, wave = tid >> 5, lane = tid & 31;
  __shared__ float gs[1024];
  __shared__ float wts[40];
  __shared__ float red[9];
  const float scale = 0.9999950000374997f;  // 1/sqrt(1+1e-5)
  for (int e = tid; e < 1024; e += 256) {
    float g = gpre[(size_t)b * 1024 + e] + gb[e];
    if (hasBN) g = g * scale * bngg[e] + bngb[e];
    gs[e] = tanhf(g);
  }
  __syncthreads();
  for (int r = wave; r < G; r += 8) {
    float mg = hasBN ? bnlg[r] * scale : 1.0f;
    float mb = hasBN ? bnlb[r] : 0.0f;
    float acc = 0.f;
    for (int e = lane; e < 1024; e += 32) {
      float l = tanhf((lpre[((size_t)b * G + r) * 1024 + e] + lb[e]) * mg + mb);
      acc += l * gs[e] * cw[e];
    }
    for (int o = 16; o > 0; o >>= 1) acc += __shfl_xor(acc, o);
    if (lane == 0) wts[r] = acc + cb[0];
  }
  __syncthreads();
  if (tid == 0) {
    float mx = -1e30f;
    for (int r = 0; r < G; ++r) mx = fmaxf(mx, wts[r]);
    float s = 0.f;
    for (int r = 0; r < G; ++r) { wts[r] = expf(wts[r] - mx); s += wts[r]; }
    for (int r = 0; r < G; ++r) wts[r] /= s;
  }
  __syncthreads();
  float v[4];
  float ss = 0.f;
#pragma unroll
  for (int p = 0; p < 4; ++p) {
    int e = tid + p * 256;
    float acc = 0.f;
    for (int r = 0; r < G; ++r) acc += wts[r] * local[((size_t)b * G + r) * 1024 + e];
    v[p] = acc;
    ss += acc * acc;
  }
  for (int o = 16; o > 0; o >>= 1) ss += __shfl_xor(ss, o);
  if (lane == 0) red[wave] = ss;
  __syncthreads();
  if (tid == 0) {
    float t = 0.f;
    for (int w8 = 0; w8 < 8; ++w8) t += red[w8];
    red[8] = t;
  }
  __syncthreads();
  float inv = 1.f / (sqrtf(red[8]) + 1e-8f);
#pragma unroll
  for (int p = 0; p < 4; ++p) out[(size_t)b * 1024 + tid + p * 256] = v[p] * inv;
}

__global__ void simgm_kernel(const float* __restrict__ ig, const float* __restrict__ cg,
                             float* __restrict__ out1) {
  int id = blockIdx.x * 256 + threadIdx.x;
  if (id >= 128 * 128) return;
  int i = id >> 7, j = id & 127;
  float acc = 0.f;
  for (int e = 0; e < 1024; ++e) acc += ig[i * 1024 + e] * cg[j * 1024 + e];
  out1[id] = acc;
}

// sim_glo: embG[j,i,:] = l2norm((img_glo[i]-cap_glo[j])^2 @ glo_w^T + glo_b). 16 i's per block.
__global__ __launch_bounds__(256) void simglo_kernel(
    const float* __restrict__ img_glo, const float* __restrict__ cap_glo,
    const _Float16* __restrict__ glow16, const float* __restrict__ glob,
    float* __restrict__ embG) {
  __shared__ _Float16 sA[16 * 32];
  __shared__ float Cst[16 * 256];
  __shared__ float nrm[16];
  const int j = blockIdx.y, i0 = blockIdx.x * 16;
  const int tid = threadIdx.x, wave = tid >> 5, lane = tid & 31;
  const int n0 = wave * 32;
  v8f a0 = {}, a1 = {};
  for (int k0 = 0; k0 < 1024; k0 += 32) {
    for (int v = tid; v < 512; v += 256) {
      int il = v >> 5, c = v & 31;
      float d = img_glo[(i0 + il) * 1024 + k0 + c] - cap_glo[j * 1024 + k0 + c];
      sA[v] = (_Float16)(d * d);
    }
    __syncthreads();
    v16h af = load_a16(sA, 32);
    v16h b0 = load_b16(glow16 + (size_t)n0 * 1024 + k0, 1024);
    v16h b1 = load_b16(glow16 + (size_t)(n0 + 16) * 1024 + k0, 1024);
    a0 = WMMAF16(af, b0, a0);
    a1 = WMMAF16(af, b1, a1);
    __syncthreads();
  }
  {
    const int col = lane & 15, rb = (lane >> 4) * 8;
#pragma unroll
    for (int r = 0; r < 8; ++r) {
      Cst[(rb + r) * 256 + n0 + col] = a0[r] + glob[n0 + col];
      Cst[(rb + r) * 256 + n0 + 16 + col] = a1[r] + glob[n0 + 16 + col];
    }
  }
  __syncthreads();
  for (int row = wave; row < 16; row += 8) {
    float ss = 0.f;
    for (int d = lane; d < 256; d += 32) { float x = Cst[row * 256 + d]; ss += x * x; }
    for (int o = 16; o > 0; o >>= 1) ss += __shfl_xor(ss, o);
    if (lane == 0) nrm[row] = ss;
  }
  __syncthreads();
  for (int v = tid; v < 4096; v += 256) {
    int il = v >> 8, d = v & 255;
    embG[((size_t)j * 128 + i0 + il) * 256 + d] = Cst[v] / (sqrtf(nrm[il]) + 1e-8f);
  }
}

// ---------------------------------------------------------------------------
// ICR small GEMM: dst[11x256] = op(src[11x256] @ W[256x256]^T + bias). Rows padded to 16.
// ---------------------------------------------------------------------------
__device__ void icr_gemm(float* dst, const float* src, const _Float16* __restrict__ W,
                         const float* __restrict__ bias, _Float16* ast, bool relu) {
  const int tid = threadIdx.x, wave = tid >> 5, lane = tid & 31;
  const int n0 = wave * 32;
  v8f a0 = {}, a1 = {};
  for (int k0 = 0; k0 < 256; k0 += 32) {
    __syncthreads();
    for (int v = tid; v < 512; v += 256) {
      int row = v >> 5, c = v & 31;
      float x = (row < 11) ? src[row * 256 + k0 + c] : 0.f;
      ast[v] = (_Float16)x;
    }
    __syncthreads();
    v16h af = load_a16(ast, 32);
    v16h b0 = load_b16(W + (size_t)n0 * 256 + k0, 256);
    v16h b1 = load_b16(W + (size_t)(n0 + 16) * 256 + k0, 256);
    a0 = WMMAF16(af, b0, a0);
    a1 = WMMAF16(af, b1, a1);
  }
  __syncthreads();
  const int col = lane & 15, rb = (lane >> 4) * 8;
#pragma unroll
  for (int r = 0; r < 8; ++r) {
    int row = rb + r;
    if (row < 11) {
      float v0 = a0[r] + bias[n0 + col];
      float v1 = a1[r] + bias[n0 + 16 + col];
      if (relu) { v0 = fmaxf(v0, 0.f); v1 = fmaxf(v1, 0.f); }
      dst[row * 256 + n0 + col] = v0;
      dst[row * 256 + n0 + 16 + col] = v1;
    }
  }
  __syncthreads();
}

// ---------------------------------------------------------------------------
// Fused per-(j,i) vector-reasoning kernel: attention -> w_ctx -> sim_loc (WMMA)
// -> emb assembly -> 3 ICR steps (WMMA) -> sigmoid dot -> sim_vr[i,j].
// ---------------------------------------------------------------------------
__global__ __launch_bounds__(256) void fused_vr_kernel(
    const float* __restrict__ Qd, const float* __restrict__ img,
    const float* __restrict__ cap, const _Float16* __restrict__ locw16,
    const float* __restrict__ locb, const float* __restrict__ embG,
    const _Float16* __restrict__ icrq16, const float* __restrict__ icrqb,
    const _Float16* __restrict__ icrk16, const float* __restrict__ icrkb,
    const _Float16* __restrict__ icrg16, const float* __restrict__ icrgb,
    const float* __restrict__ simw, const float* __restrict__ simb,
    float* __restrict__ out2) {
  __shared__ __align__(16) unsigned char smem[60928];
  float* emb = (float*)smem;                      // 11x256 f32 (persistent)
  _Float16* wctx16 = (_Float16*)(smem + 11264);   // 10x1024 f16 (phase P1-P2)
  float* qbuf = (float*)(smem + 11264);           // 11x256 f32 (P4, after wctx dead)
  float* edge = (float*)(smem + 22528);           // 121 f32 (P4)
  float* Cst = (float*)(smem + 31744);            // 16x256 f32 scratch
  float* kbuf = (float*)(smem + 48128);           // 11x256 f32 (P4) / attn (P0-P1)
  float* abuf = kbuf;                             // 10x36 attention weights
  _Float16* ast = (_Float16*)(smem + 59392);      // 16x32 f16 WMMA A stage
  float* misc = (float*)(smem + 60416);           // 128 f32 scratch
  const int i = blockIdx.x, j = blockIdx.y;
  const int tid = threadIdx.x, wave = tid >> 5, lane = tid & 31;

  // ---- P0: attention logits a = leaky(Q), l2norm over t, softmax(lambda*.) over r
  for (int v = tid; v < 360; v += 256) {
    float x = Qd[(size_t)(j * 10 + v / 36) * 4608 + i * 36 + (v % 36)];
    abuf[v] = (x > 0.f) ? x : 0.1f * x;
  }
  __syncthreads();
  if (tid < 36) {
    float ss = 0.f;
    for (int t = 0; t < 10; ++t) { float x = abuf[t * 36 + tid]; ss += x * x; }
    misc[tid] = 1.f / (sqrtf(ss) + 1e-8f);
  }
  __syncthreads();
  if (tid < 10) {
    float vals[36];
    float mx = -1e30f;
#pragma unroll
    for (int r = 0; r < 36; ++r) {
      float x = 9.0f * abuf[tid * 36 + r] * misc[r];
      vals[r] = x;
      mx = fmaxf(mx, x);
    }
    float s = 0.f;
#pragma unroll
    for (int r = 0; r < 36; ++r) { vals[r] = expf(vals[r] - mx); s += vals[r]; }
    float inv = 1.f / s;
#pragma unroll
    for (int r = 0; r < 36; ++r) abuf[tid * 36 + r] = vals[r] * inv;
  }
  __syncthreads();

  // ---- P1: w_ctx[t,e] = sum_r attn[t,r]*img[i,r,e]; per-t sumsq; store f16
  float wv[4][10];
#pragma unroll
  for (int p = 0; p < 4; ++p)
#pragma unroll
    for (int t = 0; t < 10; ++t) wv[p][t] = 0.f;
  for (int r = 0; r < 36; ++r) {
    float ar[10];
#pragma unroll
    for (int t = 0; t < 10; ++t) ar[t] = abuf[t * 36 + r];
#pragma unroll
    for (int p = 0; p < 4; ++p) {
      float im = img[((size_t)i * 36 + r) * 1024 + tid + p * 256];
#pragma unroll
      for (int t = 0; t < 10; ++t) wv[p][t] += ar[t] * im;
    }
  }
  {
    float ssl[10];
#pragma unroll
    for (int t = 0; t < 10; ++t) ssl[t] = 0.f;
#pragma unroll
    for (int p = 0; p < 4; ++p)
#pragma unroll
      for (int t = 0; t < 10; ++t) ssl[t] += wv[p][t] * wv[p][t];
#pragma unroll
    for (int t = 0; t < 10; ++t) {
      float x = ssl[t];
      for (int o = 16; o > 0; o >>= 1) x += __shfl_xor(x, o);
      if (lane == 0) misc[48 + wave * 10 + t] = x;
    }
#pragma unroll
    for (int p = 0; p < 4; ++p)
#pragma unroll
      for (int t = 0; t < 10; ++t) wctx16[t * 1024 + tid + p * 256] = (_Float16)wv[p][t];
  }
  __syncthreads();
  if (tid < 10) {
    float s = 0.f;
    for (int w8 = 0; w8 < 8; ++w8) s += misc[48 + w8 * 10 + tid];
    misc[36 + tid] = s;  // tss[t]
  }
  __syncthreads();

  // ---- P2: sim_loc = l2norm((w_ctx/nrm - cap_j)^2 @ loc_w^T + loc_b)  (WMMA, K=1024)
  {
    const int n0 = wave * 32;
    v8f a0 = {}, a1 = {};
    for (int k0 = 0; k0 < 1024; k0 += 32) {
      for (int v = tid; v < 512; v += 256) {
        int t = v >> 5, c = v & 31;
        float x = 0.f;
        if (t < 10) {
          float wn = (float)wctx16[t * 1024 + k0 + c] / (sqrtf(misc[36 + t]) + 1e-8f);
          float d = wn - cap[((size_t)j * 10 + t) * 1024 + k0 + c];
          x = d * d;
        }
        ast[v] = (_Float16)x;
      }
      __syncthreads();
      v16h af = load_a16(ast, 32);
      v16h b0 = load_b16(locw16 + (size_t)n0 * 1024 + k0, 1024);
      v16h b1 = load_b16(locw16 + (size_t)(n0 + 16) * 1024 + k0, 1024);
      a0 = WMMAF16(af, b0, a0);
      a1 = WMMAF16(af, b1, a1);
      __syncthreads();
    }
    const int col = lane & 15, rb = (lane >> 4) * 8;
#pragma unroll
    for (int r = 0; r < 8; ++r) {
      Cst[(rb + r) * 256 + n0 + col] = a0[r] + locb[n0 + col];
      Cst[(rb + r) * 256 + n0 + 16 + col] = a1[r] + locb[n0 + 16 + col];
    }
  }
  __syncthreads();
  for (int row = wave; row < 10; row += 8) {
    float ss = 0.f;
    for (int d = lane; d < 256; d += 32) { float x = Cst[row * 256 + d]; ss += x * x; }
    for (int o = 16; o > 0; o >>= 1) ss += __shfl_xor(ss, o);
    if (lane == 0) misc[48 + row] = ss;
  }
  __syncthreads();
  for (int v = tid; v < 2560; v += 256) {
    int t = v >> 8;
    emb[256 + v] = Cst[v] / (sqrtf(misc[48 + t]) + 1e-8f);  // rows 1..10
  }
  emb[tid] = embG[((size_t)j * 128 + i) * 256 + tid];  // row 0 = sim_glo
  __syncthreads();

  // ---- P4: 3 ICR steps, all in LDS
  for (int kk = 0; kk < 3; ++kk) {
    icr_gemm(qbuf, emb, icrq16 + kk * 65536, icrqb + kk * 256, ast, false);
    icr_gemm(kbuf, emb, icrk16 + kk * 65536, icrkb + kk * 256, ast, false);
    for (int pp = wave; pp < 121; pp += 8) {
      int n = pp / 11, m = pp % 11;
      float acc = 0.f;
      for (int d = lane; d < 256; d += 32) acc += qbuf[n * 256 + d] * kbuf[m * 256 + d];
      for (int o = 16; o > 0; o >>= 1) acc += __shfl_xor(acc, o);
      if (lane == 0) edge[pp] = acc;
    }
    __syncthreads();
    if (tid < 11) {
      float mx = -1e30f;
      for (int m = 0; m < 11; ++m) mx = fmaxf(mx, edge[tid * 11 + m]);
      float s = 0.f;
      float ex[11];
      for (int m = 0; m < 11; ++m) { ex[m] = expf(edge[tid * 11 + m] - mx); s += ex[m]; }
      float inv = 1.f / s;
      for (int m = 0; m < 11; ++m) edge[tid * 11 + m] = ex[m] * inv;
    }
    __syncthreads();
    for (int v = tid; v < 2816; v += 256) {  // h = edge @ emb -> kbuf
      int n = v >> 8, d = v & 255;
      float s = 0.f;
#pragma unroll
      for (int m = 0; m < 11; ++m) s += edge[n * 11 + m] * emb[m * 256 + d];
      kbuf[v] = s;
    }
    __syncthreads();
    icr_gemm(emb, kbuf, icrg16 + kk * 65536, icrgb + kk * 256, ast, true);
  }

  // ---- P5: sigmoid(emb[0] . sim_w + sim_b)
  {
    float* red = Cst;
    red[tid] = emb[tid] * simw[tid];
    __syncthreads();
    for (int s = 128; s > 0; s >>= 1) {
      if (tid < s) red[tid] += red[tid + s];
      __syncthreads();
    }
    if (tid == 0) out2[i * 128 + j] = 1.f / (1.f + expf(-(red[0] + simb[0])));
  }
}

// ---------------------------------------------------------------------------
extern "C" void kernel_launch(void* const* d_in, const int* in_sizes, int n_in,
                              void* d_out, int out_size, void* d_ws, size_t ws_size,
                              hipStream_t stream) {
  (void)in_sizes; (void)n_in; (void)out_size; (void)ws_size;
  const float* img  = (const float*)d_in[0];
  const float* cap  = (const float*)d_in[1];
  const float* vlocw = (const float*)d_in[3];
  const float* vlocb = (const float*)d_in[4];
  const float* vbnlg = (const float*)d_in[5];
  const float* vbnlb = (const float*)d_in[6];
  const float* vglow = (const float*)d_in[7];
  const float* vglob = (const float*)d_in[8];
  const float* vbngg = (const float*)d_in[9];
  const float* vbngb = (const float*)d_in[10];
  const float* vcomw = (const float*)d_in[11];
  const float* vcomb = (const float*)d_in[12];
  const float* tlocw = (const float*)d_in[13];
  const float* tlocb = (const float*)d_in[14];
  const float* tglow = (const float*)d_in[15];
  const float* tglob = (const float*)d_in[16];
  const float* tcomw = (const float*)d_in[17];
  const float* tcomb = (const float*)d_in[18];
  const float* locw  = (const float*)d_in[19];
  const float* locb  = (const float*)d_in[20];
  const float* glow  = (const float*)d_in[21];
  const float* glob  = (const float*)d_in[22];
  const float* simw  = (const float*)d_in[23];
  const float* simb  = (const float*)d_in[24];
  const float* icrqw = (const float*)d_in[25];
  const float* icrqb = (const float*)d_in[26];
  const float* icrkw = (const float*)d_in[27];
  const float* icrkb = (const float*)d_in[28];
  const float* icrgw = (const float*)d_in[29];
  const float* icrgb = (const float*)d_in[30];

  unsigned char* wsp = (unsigned char*)d_ws;
  size_t off = 0;
  auto carve = [&](size_t bytes) -> void* {
    void* p = wsp + off;
    off += (bytes + 255) & ~(size_t)255;
    return p;
  };
  float*    Qd      = (float*)carve((size_t)1280 * 4608 * 4);
  _Float16* img16   = (_Float16*)carve((size_t)4608 * 1024 * 2);
  _Float16* cap16   = (_Float16*)carve((size_t)1280 * 1024 * 2);
  _Float16* vloc16  = (_Float16*)carve((size_t)1024 * 1024 * 2);
  _Float16* vglo16  = (_Float16*)carve((size_t)1024 * 1024 * 2);
  _Float16* tloc16  = (_Float16*)carve((size_t)1024 * 1024 * 2);
  _Float16* tglo16  = (_Float16*)carve((size_t)1024 * 1024 * 2);
  _Float16* locw16  = (_Float16*)carve((size_t)256 * 1024 * 2);
  _Float16* glow16  = (_Float16*)carve((size_t)256 * 1024 * 2);
  _Float16* icrq16  = (_Float16*)carve((size_t)3 * 65536 * 2);
  _Float16* icrk16  = (_Float16*)carve((size_t)3 * 65536 * 2);
  _Float16* icrg16  = (_Float16*)carve((size_t)3 * 65536 * 2);
  _Float16* imean16 = (_Float16*)carve((size_t)128 * 1024 * 2);
  _Float16* cmean16 = (_Float16*)carve((size_t)128 * 1024 * 2);
  float*    lv      = (float*)carve((size_t)4608 * 1024 * 4);
  float*    lt      = (float*)carve((size_t)1280 * 1024 * 4);
  float*    gv      = (float*)carve((size_t)128 * 1024 * 4);
  float*    gt      = (float*)carve((size_t)128 * 1024 * 4);
  float*    imgglo  = (float*)carve((size_t)128 * 1024 * 4);
  float*    capglo  = (float*)carve((size_t)128 * 1024 * 4);
  float*    embG    = (float*)carve((size_t)128 * 128 * 256 * 4);

  float* out0 = (float*)d_out;
  float* out1 = out0 + 128 * 128 * 10;
  float* out2 = out1 + 128 * 128;

  auto cvt = [&](const float* s, _Float16* d, int n) {
    f32_to_f16<<<(n + 255) / 256, 256, 0, stream>>>(s, d, n);
  };
  cvt(img, img16, 4608 * 1024);
  cvt(cap, cap16, 1280 * 1024);
  cvt(vlocw, vloc16, 1024 * 1024);
  cvt(vglow, vglo16, 1024 * 1024);
  cvt(tlocw, tloc16, 1024 * 1024);
  cvt(tglow, tglo16, 1024 * 1024);
  cvt(locw, locw16, 256 * 1024);
  cvt(glow, glow16, 256 * 1024);
  cvt(icrqw, icrq16, 3 * 65536);
  cvt(icrkw, icrk16, 3 * 65536);
  cvt(icrgw, icrg16, 3 * 65536);

  mean_g<<<512, 256, 0, stream>>>(img, imean16, 36);
  mean_g<<<512, 256, 0, stream>>>(cap, cmean16, 10);

  // Q[jt, ir] = cap . img
  gemm_f16_nt<<<dim3(36, 20), 256, 0, stream>>>(cap16, img16, Qd, 1280, 4608, 1024);
  // SA linear layers (bias applied in sa_kernel)
  gemm_f16_nt<<<dim3(8, 72), 256, 0, stream>>>(img16, vloc16, lv, 4608, 1024, 1024);
  gemm_f16_nt<<<dim3(8, 2), 256, 0, stream>>>(imean16, vglo16, gv, 128, 1024, 1024);
  gemm_f16_nt<<<dim3(8, 20), 256, 0, stream>>>(cap16, tloc16, lt, 1280, 1024, 1024);
  gemm_f16_nt<<<dim3(8, 2), 256, 0, stream>>>(cmean16, tglo16, gt, 128, 1024, 1024);

  simlm_kernel<<<640, 256, 0, stream>>>(Qd, out0);

  sa_kernel<<<128, 256, 0, stream>>>(img, lv, gv, vlocb, vglob, vbnlg, vbnlb, vbngg, vbngb,
                                     vcomw, vcomb, imgglo, 36, 1);
  sa_kernel<<<128, 256, 0, stream>>>(cap, lt, gt, tlocb, tglob, nullptr, nullptr, nullptr,
                                     nullptr, tcomw, tcomb, capglo, 10, 0);

  simgm_kernel<<<64, 256, 0, stream>>>(imgglo, capglo, out1);

  simglo_kernel<<<dim3(8, 128), 256, 0, stream>>>(imgglo, capglo, glow16, glob, embG);

  fused_vr_kernel<<<dim3(128, 128), 256, 0, stream>>>(
      Qd, img, cap, locw16, locb, embG, icrq16, icrqb, icrk16, icrkb, icrg16, icrgb,
      simw, simb, out2);
}
